// _LoRA_qkv_32925219291175
// MI455X (gfx1250) — matrix operand order repaired and checked by reading, not compiled, not run on hardware
//
#include <hip/hip_runtime.h>
#include <hip/hip_bf16.h>

// Problem constants (from reference): x[8,64,64,1280] -> out[8,64,64,3840]
#define M_TOT   32768   // 8*64*64 tokens
#define K_TOT   1280    // C
#define N_TOT   3840    // 3*C
#define RANK    8

// GEMM tiling
#define MT      128     // macro-tile M
#define NT      128     // macro-tile N
#define KB      64      // K step = two bf16 WMMA k-slices per stage
#define NKSTEP  (K_TOT / KB)   // 20 stages
#define LDS_STRIDE 72   // KB + 8 shorts padding (144B rows -> conflict-free frag reads)
#define B_BUF_BYTES (NT * LDS_STRIDE * 2)   // 18432 B per B buffer

typedef __attribute__((ext_vector_type(16))) __bf16 v16bf;
typedef __attribute__((ext_vector_type(2)))  __bf16 v2bf;
typedef __attribute__((ext_vector_type(8)))  float  v8f;
typedef __attribute__((ext_vector_type(2)))  float  v2f;

union FragAB {
    uint4 q[2];   // 32 bytes
    v16bf v;      // 16 bf16 = 32 bytes (8 VGPRs)
};

// hardware bf16 convert (v_cvt_pk_bf16_f32), RNE
__device__ __forceinline__ unsigned int cvt_pk_bf16(float lo, float hi) {
    v2f f = {lo, hi};
    v2bf b = __builtin_convertvector(f, v2bf);
    union { v2bf b; unsigned int u; } u; u.b = b;
    return u.u;
}

__device__ __forceinline__ unsigned short f32_to_bf16_rne(float f) {
    union { float f; unsigned int u; } v; v.f = f;
    unsigned int u = v.u;
    u += 0x7FFFu + ((u >> 16) & 1u);
    return (unsigned short)(u >> 16);
}

// Async DMA: copy 64 contiguous bytes global -> LDS per lane (4x b128).
// INST_OFFSET applies to both LDS and global addresses (ISA 10.x async ops),
// and each lane's 64B chunk is contiguous on both sides, so offsets work.
__device__ __forceinline__ void async_copy_64B(unsigned lds_addr,
                                               unsigned long long gaddr) {
    asm volatile(
        "global_load_async_to_lds_b128 %0, %1, off\n\t"
        "global_load_async_to_lds_b128 %0, %1, off offset:16\n\t"
        "global_load_async_to_lds_b128 %0, %1, off offset:32\n\t"
        "global_load_async_to_lds_b128 %0, %1, off offset:48"
        :: "v"(lds_addr), "v"(gaddr) : "memory");
}

__device__ __forceinline__ void wait_async_done() {
    asm volatile("s_wait_asynccnt 0x0" ::: "memory");
}

// ---------------------------------------------------------------------------
// Kernel 1: fold LoRA into the base weight (rank-8 -> ~78 MFLOP total).
//   W_eff[o, c] = W_qkv[o, c] + s_seg * sum_r B_seg[o%C, r] * A_seg[r, c]
// stored as bf16 [N_TOT, K_TOT] row-major in workspace (9.4 MB, L2-resident).
// ---------------------------------------------------------------------------
__global__ void build_weff_kernel(const float* __restrict__ W_qkv,
                                  const float* __restrict__ A_q, const float* __restrict__ B_q,
                                  const float* __restrict__ A_k, const float* __restrict__ B_k,
                                  const float* __restrict__ A_v, const float* __restrict__ B_v,
                                  const float* __restrict__ Wq,  const float* __restrict__ Wk,
                                  const float* __restrict__ Wv,
                                  unsigned short* __restrict__ Weff) {
    int idx = blockIdx.x * blockDim.x + threadIdx.x;
    if (idx >= N_TOT * K_TOT) return;
    int o = idx / K_TOT;        // output channel in [0, 3C)
    int c = idx - o * K_TOT;    // input channel in [0, C)
    int seg = o / K_TOT;        // 0=q, 1=k, 2=v
    int ol  = o - seg * K_TOT;  // channel within segment

    const float* A; const float* Bm; float s;
    if (seg == 0)      { A = A_q; Bm = B_q; s = Wq[0]; }
    else if (seg == 1) { A = A_k; Bm = B_k; s = Wk[0]; }
    else               { A = A_v; Bm = B_v; s = Wv[0]; }

    float acc = W_qkv[(size_t)o * K_TOT + c];
    #pragma unroll
    for (int r = 0; r < RANK; ++r)
        acc += s * Bm[ol * RANK + r] * A[r * K_TOT + c];

    Weff[(size_t)o * K_TOT + c] = f32_to_bf16_rne(acc);
}

// ---------------------------------------------------------------------------
// Kernel 2: out[M, N] = x[M, K] (f32, cast bf16) @ Weff[N, K]^T + b[N]
// 256 threads = 8 waves; macro-tile 128x128; waves 4(M) x 2(N); each wave
// computes 32x64 via 2x4 accumulators; KB=64 -> 16 WMMAs per barrier.
// Double-buffered LDS. A tile: global->VGPR->cvt_pk_bf16->LDS (needs convert).
// B tile: GLOBAL_LOAD_ASYNC_TO_LDS_B128 DMA (already bf16, no VGPR round-trip),
// tracked with ASYNCcnt and fenced with s_wait_asynccnt before the barrier.
// ---------------------------------------------------------------------------
__global__ void __launch_bounds__(256)
lora_qkv_gemm_kernel(const float* __restrict__ x,
                     const unsigned short* __restrict__ Weff,
                     const float* __restrict__ bias,
                     float* __restrict__ out) {
    __shared__ unsigned short As[2][MT * LDS_STRIDE];  // 2 x 18 KB
    __shared__ unsigned short Bs[2][NT * LDS_STRIDE];  // 2 x 18 KB

    const int tid  = threadIdx.x;
    const int lane = tid & 31;
    const int wave = tid >> 5;
    const int waveM = wave & 3;         // 0..3 -> M sub-tile of 32
    const int waveN = wave >> 2;        // 0..1 -> N sub-tile of 64

    const int n0 = blockIdx.x * NT;     // 30 tiles over N
    const int m0 = blockIdx.y * MT;     // 256 tiles over M

    // cooperative-load indices: 2 threads per row, 32 elements each
    const int ldRow = tid >> 1;          // 0..127
    const int ldCol = (tid & 1) * 32;    // element offset within KB

    const float* aSrc = x + (size_t)(m0 + ldRow) * K_TOT + ldCol;
    const unsigned short* bSrc = Weff + (size_t)(n0 + ldRow) * K_TOT + ldCol;

    // LDS byte address (low 32 bits of flat address) of this thread's B slot
    const unsigned bLds0 =
        (unsigned)(size_t)(void*)&Bs[0][ldRow * LDS_STRIDE + ldCol];

    v8f acc[2][4];
    #pragma unroll
    for (int mt = 0; mt < 2; ++mt)
        #pragma unroll
        for (int nt = 0; nt < 4; ++nt)
            acc[mt][nt] = (v8f){0.f, 0.f, 0.f, 0.f, 0.f, 0.f, 0.f, 0.f};

    const int lr = lane & 15;           // row/col within a 16-tile
    const int lh = lane >> 4;           // lane half selects K sub-block

    float4 aReg[8];

    // ---- prologue: stage 0 -> buffer 0 ----
    async_copy_64B(bLds0, (unsigned long long)bSrc);
    #pragma unroll
    for (int i = 0; i < 8; ++i) aReg[i] = *(const float4*)(aSrc + i * 4);
    {
        unsigned int* da = (unsigned int*)&As[0][ldRow * LDS_STRIDE + ldCol];
        #pragma unroll
        for (int i = 0; i < 4; ++i) {
            uint4 p = { cvt_pk_bf16(aReg[2*i].x, aReg[2*i].y),
                        cvt_pk_bf16(aReg[2*i].z, aReg[2*i].w),
                        cvt_pk_bf16(aReg[2*i+1].x, aReg[2*i+1].y),
                        cvt_pk_bf16(aReg[2*i+1].z, aReg[2*i+1].w) };
            *(uint4*)(da + i * 4) = p;
        }
    }
    wait_async_done();
    __syncthreads();

    for (int it = 0; it < NKSTEP; ++it) {
        const int buf = it & 1;
        const int nbuf = buf ^ 1;

        // ---- kick off stage it+1: B via async DMA, A into registers ----
        if (it + 1 < NKSTEP) {
            const int kn = (it + 1) * KB;
            async_copy_64B(bLds0 + (unsigned)nbuf * B_BUF_BYTES,
                           (unsigned long long)(bSrc + kn));
            #pragma unroll
            for (int i = 0; i < 8; ++i)
                aReg[i] = *(const float4*)(aSrc + kn + i * 4);
        }

        // ---- compute: two 16x16x32 k-slices from LDS buffer `buf` ----
        #pragma unroll
        for (int ks = 0; ks < 2; ++ks) {
            FragAB afrag[2];
            #pragma unroll
            for (int mt = 0; mt < 2; ++mt) {
                const unsigned short* rowp =
                    &As[buf][(waveM * 32 + mt * 16 + lr) * LDS_STRIDE + ks * 32];
                // VGPR0-3: K = lh*8 + 0..7 ; VGPR4-7: K = 16 + lh*8 + 0..7
                afrag[mt].q[0] = *(const uint4*)(rowp + lh * 8);
                afrag[mt].q[1] = *(const uint4*)(rowp + 16 + lh * 8);
            }
            FragAB bfrag[4];
            #pragma unroll
            for (int nt = 0; nt < 4; ++nt) {
                const unsigned short* rowp =
                    &Bs[buf][(waveN * 64 + nt * 16 + lr) * LDS_STRIDE + ks * 32];
                // VGPR0-3: K = lh*8 + 0..7 ; VGPR4-7: K = 16 + lh*8 + 0..7
                bfrag[nt].q[0] = *(const uint4*)(rowp + lh * 8);
                bfrag[nt].q[1] = *(const uint4*)(rowp + 16 + lh * 8);
            }
            #pragma unroll
            for (int mt = 0; mt < 2; ++mt)
                #pragma unroll
                for (int nt = 0; nt < 4; ++nt)
                    acc[mt][nt] = __builtin_amdgcn_wmma_f32_16x16x32_bf16(
                        false, afrag[mt].v,
                        false, bfrag[nt].v,
                        (short)0, acc[mt][nt],
                        false, false);
        }

        // ---- commit A for stage it+1 into the other buffer ----
        if (it + 1 < NKSTEP) {
            unsigned int* da = (unsigned int*)&As[nbuf][ldRow * LDS_STRIDE + ldCol];
            #pragma unroll
            for (int i = 0; i < 4; ++i) {
                uint4 p = { cvt_pk_bf16(aReg[2*i].x, aReg[2*i].y),
                            cvt_pk_bf16(aReg[2*i].z, aReg[2*i].w),
                            cvt_pk_bf16(aReg[2*i+1].x, aReg[2*i+1].y),
                            cvt_pk_bf16(aReg[2*i+1].z, aReg[2*i+1].w) };
                *(uint4*)(da + i * 4) = p;
            }
            wait_async_done();   // B DMA for stage it+1 landed in LDS
        }
        __syncthreads();
    }

    // ---- epilogue: add bias, non-temporal f32 stores (keep L2 for x/Weff) ----
    // C/D layout: VGPR r -> lanes 0-15: M=r, N=lane ; lanes 16-31: M=r+8, N=lane-16
    #pragma unroll
    for (int mt = 0; mt < 2; ++mt) {
        #pragma unroll
        for (int nt = 0; nt < 4; ++nt) {
            const int nGlob = n0 + waveN * 64 + nt * 16 + lr;
            const float b = bias[nGlob];
            const int mBase = m0 + waveM * 32 + mt * 16 + lh * 8;
            #pragma unroll
            for (int r = 0; r < 8; ++r) {
                __builtin_nontemporal_store(
                    acc[mt][nt][r] + b,
                    &out[(size_t)(mBase + r) * N_TOT + nGlob]);
            }
        }
    }
}

// ---------------------------------------------------------------------------
// Launch
// ---------------------------------------------------------------------------
extern "C" void kernel_launch(void* const* d_in, const int* in_sizes, int n_in,
                              void* d_out, int out_size, void* d_ws, size_t ws_size,
                              hipStream_t stream) {
    const float* x     = (const float*)d_in[0];
    const float* W_qkv = (const float*)d_in[1];
    const float* b_qkv = (const float*)d_in[2];
    const float* A_q   = (const float*)d_in[3];
    const float* B_q   = (const float*)d_in[4];
    const float* A_k   = (const float*)d_in[5];
    const float* B_k   = (const float*)d_in[6];
    const float* A_v   = (const float*)d_in[7];
    const float* B_v   = (const float*)d_in[8];
    const float* Wq    = (const float*)d_in[9];
    const float* Wk    = (const float*)d_in[10];
    const float* Wv    = (const float*)d_in[11];
    float* out = (float*)d_out;

    unsigned short* Weff = (unsigned short*)d_ws;  // 3840*1280*2 B = 9.4 MB

    // 1) fold LoRA into weights (bf16)
    {
        int total = N_TOT * K_TOT;
        int blocks = (total + 255) / 256;
        build_weff_kernel<<<blocks, 256, 0, stream>>>(
            W_qkv, A_q, B_q, A_k, B_k, A_v, B_v, Wq, Wk, Wv, Weff);
    }

    // 2) fused GEMM + bias
    {
        dim3 grid(N_TOT / NT, M_TOT / MT);  // (30, 256)
        lora_qkv_gemm_kernel<<<grid, 256, 0, stream>>>(x, Weff, b_qkv, out);
    }
}
